// SparseCausalSelfAttention_35708358099050
// MI455X (gfx1250) — matrix operand order repaired and checked by reading, not compiled, not run on hardware
//
#include <hip/hip_runtime.h>
#include <hip/hip_bf16.h>

typedef __attribute__((ext_vector_type(16))) _Float16 v16h;
typedef __attribute__((ext_vector_type(8)))  float    v8f;

union Frag { v16h h; unsigned int u[8]; };

#define LDS_WAIT()   asm volatile("s_wait_dscnt 0x0" ::: "memory")
#define ASYNC_WAIT() asm volatile("s_wait_asynccnt 0x0" ::: "memory")

constexpr int BB   = 2;
constexpr int TT   = 1024;
constexpr int KTOP = 2;
constexpr int HH   = 8;
constexpr int DD   = 64;

// pack two f32 -> one dword of two f16
__device__ __forceinline__ unsigned pkh(float a, float b) {
    union { _Float16 h[2]; unsigned u; } t;
    t.h[0] = (_Float16)a;
    t.h[1] = (_Float16)b;
    return t.u;
}

// One wave32 handles one (b, kk, h, 16-row block) task.
// Column blocks of 32 are processed right-to-left so the strictly-triangular
// suffix sum of log_beta is carried in registers.  K/V tiles are staged into
// LDS with CDNA5 async copies (ASYNCcnt), WMMA f16->f32 does both matmuls.
__global__ __launch_bounds__(128, 1)
void fox_attn_wmma(const float* __restrict__ qg,
                   const float* __restrict__ kg,
                   const float* __restrict__ vg,
                   float* __restrict__ og)
{
    __shared__ __align__(16) float sKall[4][32 * 64]; // K tile, [row][d], f32 (32KB)
    __shared__ __align__(16) float sVall[4][32 * 64]; // V tile, [row][d], f32 (32KB)

    const int lane = threadIdx.x & 31;
    const int wid  = threadIdx.x >> 5;
    const int l16  = lane & 15;
    const int g    = lane >> 4;            // 16-lane half id

    const int task = blockIdx.x * 4 + wid; // 512 blocks * 4 waves = 2048 tasks
    const int rt = task & 63;              // row tile (16 rows)
    const int h  = (task >> 6) & 7;
    const int kk = (task >> 9) & 1;
    const int b  = (task >> 10) & 1;
    const int r0 = rt << 4;

    float*     sk32 = sKall[wid];
    float*     sv32 = sVall[wid];
    _Float16*  satt = reinterpret_cast<_Float16*>(sk32); // att tile (16x32 f16 = 1KB) aliases K

    // LDS byte offsets for the async-copy destinations (low 32 bits of the
    // flat address of a shared object are the LDS byte offset on CDNA5).
    const unsigned ldsK0 = (unsigned)(uintptr_t)sk32 + (unsigned)lane * 16u;
    const unsigned ldsV0 = (unsigned)(uintptr_t)sv32 + (unsigned)lane * 16u;

    // ---- Q fragment in WMMA A-layout (16x32 f16), two K-halves for D=64 ----
    Frag aq0, aq1;
    {
        const float* qrow = qg + ((((size_t)b * TT + (r0 + l16)) * KTOP + kk) * HH + h) * DD;
        #pragma unroll
        for (int v = 0; v < 8; ++v) {
            const int d = ((v & 4) << 2) + ((v & 3) << 1) + (g << 3); // A-layout K index
            float2 f0 = *reinterpret_cast<const float2*>(qrow + d);
            float2 f1 = *reinterpret_cast<const float2*>(qrow + d + 32);
            aq0.u[v] = pkh(f0.x, f0.y);
            aq1.u[v] = pkh(f1.x, f1.y);
        }
    }

    v8f yacc[4];                         // Y tile 16x64 f32, C-layout, 4 N-slices
    #pragma unroll
    for (int ds = 0; ds < 4; ++ds) yacc[ds] = (v8f){};
    float rsum[8] = {0, 0, 0, 0, 0, 0, 0, 0}; // per-row suffix carry

    const float* kbase = kg + (((size_t)b * TT) * HH + h) * DD;
    const float* vbase = vg + (((size_t)b * TT) * HH + h) * DD;

    const int cbmax = (r0 + 47) >> 5;    // ceil((r0+16)/32) 32-col blocks
    for (int cb = cbmax - 1; cb >= 0; --cb) {
        const int c0 = cb << 5;

        // ---- async-copy K and V tiles (32 rows x 64 f32 each) into LDS ----
        LDS_WAIT();   // prior iteration's LDS reads must retire before overwrite
        {
            // lane covers 4 consecutive f32 (16B); rows advance 2 per step.
            unsigned long long gk = (unsigned long long)(uintptr_t)
                (kbase + (size_t)(c0 + g) * (HH * DD)) + (unsigned)l16 * 16u;
            unsigned long long gv = (unsigned long long)(uintptr_t)
                (vbase + (size_t)(c0 + g) * (HH * DD)) + (unsigned)l16 * 16u;
            unsigned lk = ldsK0, lv = ldsV0;
            #pragma unroll
            for (int it = 0; it < 16; ++it) {
                asm volatile("global_load_async_to_lds_b128 %0, %1, off"
                             :: "v"(lk), "v"(gk) : "memory");
                asm volatile("global_load_async_to_lds_b128 %0, %1, off"
                             :: "v"(lv), "v"(gv) : "memory");
                lk += 512u;  lv += 512u;      // 128 f32 per step
                gk += 4096u; gv += 4096u;     // 2 rows * H*D*4 bytes
            }
        }
        ASYNC_WAIT();

        // ---- S = (Q·K^T) for both 16-col subtiles (K=64 via 2 chained K=32 WMMAs) ----
        v8f s_sub[2];
        #pragma unroll
        for (int sub = 0; sub < 2; ++sub) {
            Frag bk0, bk1;
            const int jrow = sub * 16 + l16;   // B-layout: N = l16, K = table order (matches A)
            #pragma unroll
            for (int v = 0; v < 8; ++v) {
                const int d = ((v & 4) << 2) + ((v & 3) << 1) + (g << 3);
                float2 f0 = *reinterpret_cast<const float2*>(&sk32[jrow * 64 + d]);
                float2 f1 = *reinterpret_cast<const float2*>(&sk32[jrow * 64 + d + 32]);
                bk0.u[v] = pkh(f0.x, f0.y);
                bk1.u[v] = pkh(f1.x, f1.y);
            }
            v8f acc = (v8f){};
            acc = __builtin_amdgcn_wmma_f32_16x16x32_f16(false, aq0.h, false, bk0.h,
                                                         (short)0, acc, false, false);
            acc = __builtin_amdgcn_wmma_f32_16x16x32_f16(false, aq1.h, false, bk1.h,
                                                         (short)0, acc, false, false);
            s_sub[sub] = acc;
        }
        LDS_WAIT();   // all K-tile reads retired before att aliases the region

        // ---- sigmoid / log-sigmoid / suffix scan; right subtile first ----
        #pragma unroll
        for (int sub = 1; sub >= 0; --sub) {
            #pragma unroll
            for (int v = 0; v < 8; ++v) {
                const int i  = v + (g << 3);        // C-layout row
                const int ig = r0 + i;
                const int jg = c0 + sub * 16 + l16; // C-layout col
                const float x = s_sub[sub][v] * 0.125f;   // 1/sqrt(64)
                const bool keep = (jg <= ig);
                const float ex = __expf(-fabsf(x));
                float z  = keep ? ((x >= 0.0f) ? 1.0f / (1.0f + ex) : ex / (1.0f + ex)) : 0.0f;
                float lb = keep ? -(fmaxf(x, 0.0f) + log1pf(ex)) : 0.0f;   // log_sigmoid(-x)
                // inclusive suffix scan across the 16-lane segment
                float sc = lb;
                #pragma unroll
                for (int off = 1; off < 16; off <<= 1) {
                    float t = __shfl_down(sc, off, 16);
                    if (l16 + off < 16) sc += t;
                }
                const float att = z * __expf((sc - lb) + rsum[v]); // excl suffix + carry
                rsum[v] += __shfl(sc, 0, 16);                      // tile row-total -> carry
                satt[i * 32 + sub * 16 + l16] = (_Float16)att;
            }
        }
        LDS_WAIT();

        // ---- Y += att (16x32) · V (32x64) ----
        Frag aa;
        #pragma unroll
        for (int v = 0; v < 8; ++v) {
            const int j = ((v & 4) << 2) + ((v & 3) << 1) + (g << 3);  // A-layout K index
            aa.u[v] = *reinterpret_cast<const unsigned int*>(&satt[l16 * 32 + j]);
        }
        #pragma unroll
        for (int ds = 0; ds < 4; ++ds) {
            Frag bv;
            const int dcol = ds * 16 + l16;    // B-layout: N = dcol, K = table order (matches A)
            #pragma unroll
            for (int v = 0; v < 8; ++v) {
                const int c = ((v & 4) << 2) + ((v & 3) << 1) + (g << 3);
                bv.u[v] = pkh(sv32[c * 64 + dcol], sv32[(c + 1) * 64 + dcol]);
            }
            yacc[ds] = __builtin_amdgcn_wmma_f32_16x16x32_f16(false, aa.h, false, bv.h,
                                                              (short)0, yacc[ds], false, false);
        }
    }

    // ---- write Y (f32), C-layout -> (B,T,K,H,D) ----
    #pragma unroll
    for (int ds = 0; ds < 4; ++ds) {
        #pragma unroll
        for (int v = 0; v < 8; ++v) {
            const int i = v + (g << 3);
            og[((((size_t)b * TT + (r0 + i)) * KTOP + kk) * HH + h) * DD + ds * 16 + l16] =
                yacc[ds][v];
        }
    }
}

extern "C" void kernel_launch(void* const* d_in, const int* in_sizes, int n_in,
                              void* d_out, int out_size, void* d_ws, size_t ws_size,
                              hipStream_t stream) {
    (void)in_sizes; (void)n_in; (void)d_ws; (void)ws_size; (void)out_size;
    const float* q = (const float*)d_in[0];
    const float* k = (const float*)d_in[1];
    const float* v = (const float*)d_in[2];
    // d_in[3] (mask) and d_in[4] (cum_weight) are structural tril matrices; the
    // kernel implements them analytically (causal mask + suffix scan).
    float* out = (float*)d_out;

    // 2048 wave-tasks (B*K*H*T/16), 4 waves (128 threads) per block.
    fox_attn_wmma<<<dim3(512), dim3(128), 0, stream>>>(q, k, v, out);
}
